// GNNHomo_83382495084843
// MI455X (gfx1250) — matrix operand order repaired and checked by reading, not compiled, not run on hardware
//
#include <hip/hip_runtime.h>
#include <float.h>

// ---- CDNA5 WMMA types ------------------------------------------------------
typedef __attribute__((ext_vector_type(16))) __bf16 v16bf;
typedef __attribute__((ext_vector_type(8)))  float  v8f;
typedef __attribute__((ext_vector_type(4)))  unsigned int uix4;
typedef __attribute__((ext_vector_type(4)))  float  fx4;

#define HF   96     // feature width (F == H == 96)
#define KTOT 288    // fused K = 96 (self) + 96 (sd) + 96 (ds)
#define KCH  9      // KTOT / 32 k-chunks per WMMA
#define GEMM_ROWS 64
#define GEMM_THREADS 192   // 6 wave32 -> 6 column tiles of 16

union Frag { v16bf v; unsigned int u[8]; uix4 q[2]; };

__device__ __forceinline__ unsigned short f2bf(float f) {
  unsigned int u = __float_as_uint(f);
  u += 0x7FFFu + ((u >> 16) & 1u);   // round-to-nearest-even
  return (unsigned short)(u >> 16);
}

// ---- trivial utility kernels ----------------------------------------------
__global__ void k_zero_u32(unsigned int* p, int n) {
  int i = blockIdx.x * blockDim.x + threadIdx.x;
  if (i < n) p[i] = 0u;
}

__global__ void k_fill_f32(float* p, float v, int n) {
  int i = blockIdx.x * blockDim.x + threadIdx.x;
  if (i < n) p[i] = v;
}

// ---- weight packing: bf16, fragment-order, ALPHA folded in ----------------
// wpack layout: [layer][kc][jt][lane][idx] (16 bf16 per lane = 32 contiguous B)
// B fragment (16-bit, 32x16): lane half selects K in [16h, 16h+16),
// value idx within lane = K offset inside that block; column = jt*16 + lane%16.
__global__ void k_pack_w(const float* W0s, const float* W0a, const float* W0b,
                         const float* W1s, const float* W1a, const float* W1b,
                         const float* W2s, const float* W2a, const float* W2b,
                         unsigned short* wpack) {
  int d = blockIdx.x * blockDim.x + threadIdx.x;
  if (d >= 3 * KTOT * HF) return;
  const int PER = KTOT * HF;                  // 27648
  int l  = d / PER;
  int r  = d % PER;
  int kc = r / 3072;                          // 6*32*16
  int r2 = r % 3072;
  int jt   = r2 / 512;                        // 32*16
  int lane = (r2 >> 4) & 31;
  int idx  = r2 & 15;
  int half = lane >> 4;
  int col  = jt * 16 + (lane & 15);
  int k    = kc * 32 + half * 16 + idx;
  const float *Ws, *Wa, *Wb;
  if (l == 0)      { Ws = W0s; Wa = W0a; Wb = W0b; }
  else if (l == 1) { Ws = W1s; Wa = W1a; Wb = W1b; }
  else             { Ws = W2s; Wa = W2a; Wb = W2b; }
  float w;
  if (k < 96)       w = Ws[k * HF + col];                    // self
  else if (k < 192) w = 0.5f * Wa[(k - 96) * HF + col];      // (1-ALPHA)*sd
  else              w = 0.5f * Wb[(k - 192) * HF + col];     // ALPHA*ds
  wpack[d] = f2bf(w);
}

__global__ void k_pack_b(const float* b0s, const float* b0a, const float* b0b,
                         const float* b1s, const float* b1a, const float* b1b,
                         const float* b2s, const float* b2a, const float* b2b,
                         float* bcat) {
  int t = blockIdx.x * blockDim.x + threadIdx.x;
  if (t >= 3 * HF) return;
  int l = t / HF, j = t % HF;
  const float* s = (l == 0) ? b0s : (l == 1) ? b1s : b2s;
  const float* a = (l == 0) ? b0a : (l == 1) ? b1a : b2a;
  const float* b = (l == 0) ? b0b : (l == 1) ? b1b : b2b;
  bcat[t] = s[j] + 0.5f * (a[j] + b[j]);
}

// ---- degree counts (once; shared across layers) ---------------------------
__global__ void k_degree(const int* __restrict__ src, const int* __restrict__ dst,
                         int* deg_sd, int* deg_ds, int E) {
  int e = blockIdx.x * blockDim.x + threadIdx.x;
  if (e >= E) return;
  atomicAdd(&deg_sd[dst[e]], 1);
  atomicAdd(&deg_ds[src[e]], 1);
}

// ---- edge scatter: fp32 global atomics, L2-resident accumulators ----------
// 3 threads per edge, 32 features each; b128 gathers of the source rows.
__global__ void k_scatter(const float* __restrict__ x,
                          const int* __restrict__ src, const int* __restrict__ dst,
                          float* acc_sd, float* acc_ds, int E) {
  long long t = (long long)blockIdx.x * blockDim.x + threadIdx.x;
  int e = (int)(t / 3);
  int c = (int)(t % 3) * 32;
  if (e >= E) return;
  int s = src[e], d = dst[e];
  const fx4* xs = (const fx4*)(x + (long long)s * HF + c);
  const fx4* xd = (const fx4*)(x + (long long)d * HF + c);
  float* asd = acc_sd + (long long)d * HF + c;
  float* ads = acc_ds + (long long)s * HF + c;
#pragma unroll
  for (int i = 0; i < 8; ++i) {
    fx4 vs = xs[i];
    fx4 vd = xd[i];
#pragma unroll
    for (int j = 0; j < 4; ++j) {
      unsafeAtomicAdd(asd + 4 * i + j, vs[j]);   // global_atomic_add_f32
      unsafeAtomicAdd(ads + 4 * i + j, vd[j]);
    }
  }
}

// ---- fused dir-SAGE layer GEMM: [x|agg_sd|agg_ds] @ Wcat, +bias, ReLU -----
// 64-row slab in LDS (bf16), 6 waves x 4 row tiles, v_wmma_f32_16x16x32_bf16.
__global__ __launch_bounds__(GEMM_THREADS)
void k_gemm(const float* __restrict__ xin,
            const float* __restrict__ acc_sd, const float* __restrict__ acc_ds,
            const int* __restrict__ deg_sd, const int* __restrict__ deg_ds,
            const unsigned short* __restrict__ wpack,   // this layer's fragments
            const float* __restrict__ bcat,             // this layer's fused bias
            float* __restrict__ out, int N) {
  __shared__ unsigned short sA[GEMM_ROWS * KTOT];       // 36 KB
  const int rowBase = blockIdx.x * GEMM_ROWS;

  // Stage A slab as bf16; fuse the scatter-mean division here.
  for (int i = threadIdx.x * 2; i < GEMM_ROWS * KTOT; i += GEMM_THREADS * 2) {
    int rr = i / KTOT, c = i % KTOT;
    int r = rowBase + rr;
    float v0 = 0.f, v1 = 0.f;
    if (r < N) {
      if (c < 96) {
        v0 = xin[r * HF + c];
        v1 = xin[r * HF + c + 1];
      } else if (c < 192) {
        int cnt = deg_sd[r];
        float inv = 1.0f / (float)(cnt > 1 ? cnt : 1);
        v0 = acc_sd[r * HF + c - 96] * inv;
        v1 = acc_sd[r * HF + c - 95] * inv;
      } else {
        int cnt = deg_ds[r];
        float inv = 1.0f / (float)(cnt > 1 ? cnt : 1);
        v0 = acc_ds[r * HF + c - 192] * inv;
        v1 = acc_ds[r * HF + c - 191] * inv;
      }
    }
    ((unsigned int*)sA)[i >> 1] =
        (unsigned int)f2bf(v0) | ((unsigned int)f2bf(v1) << 16);
  }
  __syncthreads();

  const int wid  = threadIdx.x / 32;   // column tile jt = 0..5
  const int lane = threadIdx.x % 32;
  const int half = lane >> 4;
  const int m    = lane & 15;
  const float bias = bcat[wid * 16 + m];

  v8f acc[4] = {};   // 4 row tiles x 16x16 f32
  const unsigned int* uA = (const unsigned int*)sA;   // 144 uints per row

  for (int kc = 0; kc < KCH; ++kc) {
    Frag b;
    const uix4* bp = (const uix4*)(wpack + ((((kc * 6 + wid) * 32) + lane) << 4));
    b.q[0] = bp[0];
    b.q[1] = bp[1];
#pragma unroll
    for (int rt = 0; rt < 4; ++rt) {
      // 16-bit A 16x32 fragment: lane half -> K block 16h..16h+15 (matches B packing)
      Frag a;
      int base = (rt * 16 + m) * (KTOT / 2) + kc * 16 + half * 8;
      a.u[0] = uA[base + 0];
      a.u[1] = uA[base + 1];
      a.u[2] = uA[base + 2];
      a.u[3] = uA[base + 3];
      a.u[4] = uA[base + 4];
      a.u[5] = uA[base + 5];
      a.u[6] = uA[base + 6];
      a.u[7] = uA[base + 7];
      acc[rt] = __builtin_amdgcn_wmma_f32_16x16x32_bf16(
          false, a.v, false, b.v, (short)0, acc[rt], false, false);
    }
  }

  // Epilogue: +bias, ReLU, store. D layout: lane half -> M = 8h + vgpr.
#pragma unroll
  for (int rt = 0; rt < 4; ++rt) {
    int row0 = rowBase + rt * 16 + half * 8;
#pragma unroll
    for (int v = 0; v < 8; ++v) {
      int row = row0 + v;
      if (row < N) out[row * HF + wid * 16 + m] = fmaxf(acc[rt][v] + bias, 0.f);
    }
  }
}

// ---- global max pool via float-as-int atomic max --------------------------
__global__ void k_poolmax(const float* __restrict__ h, const int* __restrict__ batch,
                          float* pool, int N) {
  long long t = (long long)blockIdx.x * blockDim.x + threadIdx.x;
  if (t >= (long long)N * HF) return;
  int n = (int)(t / HF), f = (int)(t % HF);
  float v = h[t];
  float* addr = &pool[batch[n] * HF + f];
  int iv = __float_as_int(v);
  if (iv >= 0) atomicMax((int*)addr, iv);
  else         atomicMin((unsigned int*)addr, (unsigned int)iv);
}

// ---- final MLP: 96 -> 5 -> 1 per graph ------------------------------------
__global__ void k_mlp(const float* __restrict__ pool,
                      const float* __restrict__ w1, const float* __restrict__ b1,
                      const float* __restrict__ w2, const float* __restrict__ b2,
                      float* __restrict__ out, int G) {
  int g = blockIdx.x * blockDim.x + threadIdx.x;
  if (g >= G) return;
  float hb[5];
#pragma unroll
  for (int j = 0; j < 5; ++j) hb[j] = b1[j];
  for (int k = 0; k < HF; ++k) {
    float xv = pool[g * HF + k];
#pragma unroll
    for (int j = 0; j < 5; ++j) hb[j] += xv * w1[k * 5 + j];
  }
  float o = b2[0];
#pragma unroll
  for (int j = 0; j < 5; ++j) o += fmaxf(hb[j], 0.f) * w2[j];
  out[g] = o;
}

// ---------------------------------------------------------------------------
static inline size_t alignup(size_t x) { return (x + 255) & ~(size_t)255; }

extern "C" void kernel_launch(void* const* d_in, const int* in_sizes, int n_in,
                              void* d_out, int out_size, void* d_ws, size_t ws_size,
                              hipStream_t stream) {
  (void)n_in; (void)ws_size;
  const int N = in_sizes[0] / HF;
  const int E = in_sizes[1] / 2;
  const int G = out_size;

  const float* x     = (const float*)d_in[0];
  const int*   src   = (const int*)d_in[1];
  const int*   dst   = src + E;
  const int*   batch = (const int*)d_in[2];
  // setup_inputs dict order: per layer Wself, Wsd, Wds, bself, bsd, bds
  const float* Ws[3] = {(const float*)d_in[3],  (const float*)d_in[9],  (const float*)d_in[15]};
  const float* Wa[3] = {(const float*)d_in[4],  (const float*)d_in[10], (const float*)d_in[16]};
  const float* Wb[3] = {(const float*)d_in[5],  (const float*)d_in[11], (const float*)d_in[17]};
  const float* bs[3] = {(const float*)d_in[6],  (const float*)d_in[12], (const float*)d_in[18]};
  const float* ba[3] = {(const float*)d_in[7],  (const float*)d_in[13], (const float*)d_in[19]};
  const float* bb[3] = {(const float*)d_in[8],  (const float*)d_in[14], (const float*)d_in[20]};
  const float* lin1_w = (const float*)d_in[21];
  const float* lin1_b = (const float*)d_in[22];
  const float* lin2_w = (const float*)d_in[23];
  const float* lin2_b = (const float*)d_in[24];

  // workspace carve (~77 MB)
  char* p = (char*)d_ws;
  float* h1    = (float*)p; p += alignup((size_t)N * HF * 4);
  float* h2    = (float*)p; p += alignup((size_t)N * HF * 4);
  float* accsd = (float*)p; p += alignup((size_t)N * HF * 4);
  float* accds = (float*)p; p += alignup((size_t)N * HF * 4);
  int*   degsd = (int*)p;   p += alignup((size_t)N * 4);
  int*   degds = (int*)p;   p += alignup((size_t)N * 4);
  unsigned short* wpack = (unsigned short*)p; p += alignup((size_t)3 * KTOT * HF * 2);
  float* bcat  = (float*)p; p += alignup((size_t)3 * HF * 4);
  float* pool  = (float*)p; p += alignup((size_t)G * HF * 4);

  const int TPB = 256;

  // 1) pack weights/biases (bf16 fragment order, ALPHA folded)
  k_pack_w<<<(3 * KTOT * HF + TPB - 1) / TPB, TPB, 0, stream>>>(
      Ws[0], Wa[0], Wb[0], Ws[1], Wa[1], Wb[1], Ws[2], Wa[2], Wb[2], wpack);
  k_pack_b<<<(3 * HF + TPB - 1) / TPB, TPB, 0, stream>>>(
      bs[0], ba[0], bb[0], bs[1], ba[1], bb[1], bs[2], ba[2], bb[2], bcat);

  // 2) degrees (layer-invariant)
  k_zero_u32<<<(N + TPB - 1) / TPB, TPB, 0, stream>>>((unsigned int*)degsd, N);
  k_zero_u32<<<(N + TPB - 1) / TPB, TPB, 0, stream>>>((unsigned int*)degds, N);
  k_degree<<<(E + TPB - 1) / TPB, TPB, 0, stream>>>(src, dst, degsd, degds, E);

  // 3) three fused layers
  const float* cur = x;
  float* outs[3] = {h1, h2, h1};
  for (int l = 0; l < 3; ++l) {
    k_zero_u32<<<(N * HF + TPB - 1) / TPB, TPB, 0, stream>>>((unsigned int*)accsd, N * HF);
    k_zero_u32<<<(N * HF + TPB - 1) / TPB, TPB, 0, stream>>>((unsigned int*)accds, N * HF);
    long long sthreads = (long long)E * 3;
    k_scatter<<<(unsigned int)((sthreads + TPB - 1) / TPB), TPB, 0, stream>>>(
        cur, src, dst, accsd, accds, E);
    k_gemm<<<(N + GEMM_ROWS - 1) / GEMM_ROWS, GEMM_THREADS, 0, stream>>>(
        cur, accsd, accds, degsd, degds,
        wpack + (size_t)l * KTOT * HF, bcat + l * HF, outs[l], N);
    cur = outs[l];
  }

  // 4) global max pool + MLP head
  k_fill_f32<<<(G * HF + TPB - 1) / TPB, TPB, 0, stream>>>(pool, -FLT_MAX, G * HF);
  long long pthreads = (long long)N * HF;
  k_poolmax<<<(unsigned int)((pthreads + TPB - 1) / TPB), TPB, 0, stream>>>(
      cur, batch, pool, N);
  k_mlp<<<1, 64, 0, stream>>>(pool, lin1_w, lin1_b, lin2_w, lin2_b, (float*)d_out, G);
}
